// Insulin_CHO_sim_60387240182123
// MI455X (gfx1250) — matrix operand order repaired and checked by reading, not compiled, not run on hardware
//
#include <hip/hip_runtime.h>
#include <hip/hip_bf16.h>

// ---------------------------------------------------------------------------
// CDNA5 (gfx1250) implementation of the coupling-flow reference.
// All GEMMs run on v_wmma_f32_16x16x32_bf16 (wave32, 16x16 tiles, fp32 acc).
// Weights are converted fp32->bf16 into WMMA-friendly transposed layouts in
// d_ws each launch (deterministic, ~14MB => negligible at 23.3 TB/s).
// ---------------------------------------------------------------------------

typedef __attribute__((ext_vector_type(16))) __bf16 v16bf;
typedef __attribute__((ext_vector_type(8)))  float  v8f;

struct NetPtrs {
  unsigned short* W1T;  // [H][32]  bf16, rows k>=D zeroed (K padded to 32)
  unsigned short* W2T;  // [H][H]   bf16, transposed: [n][k]
  unsigned short* W3T;  // [16][H]  bf16, rows n>=D zeroed
  float* b1; float* b2; float* b3; // b3 padded to 16
};

__device__ __forceinline__ unsigned short f2bf(float f) {
  unsigned u = __float_as_uint(f);
  u += 0x7fffu + ((u >> 16) & 1u);          // round-to-nearest-even
  return (unsigned short)(u >> 16);
}

union Frag { v16bf v; uint4 q[2]; };

// A-fragment (16x32 bf16): lane half 0 -> K {kb..kb+7, kb+16..kb+23},
// half 1 -> K {kb+8..kb+15, kb+24..kb+31}  (ISA 7.12.2 16-bit A layout)
__device__ __forceinline__ v16bf loadA16(const unsigned short* row, int kbase, int half) {
  Frag f;
  const unsigned short* p = row + kbase + 8 * half;
  f.q[0] = *(const uint4*)(p);
  f.q[1] = *(const uint4*)(p + 16);
  return f.v;
}

// B-fragment (32x16 bf16) from transposed weights W[n][k]:
// lane half 0 -> K {kb..kb+7, kb+16..kb+23}, half 1 -> K {kb+8..kb+15, kb+24..kb+31}
__device__ __forceinline__ v16bf loadB16(const unsigned short* rowN, int kbase, int half) {
  Frag f;
  const unsigned short* p = rowN + kbase + 8 * half;
  f.q[0] = *(const uint4*)(p);
  f.q[1] = *(const uint4*)(p + 16);
  return f.v;
}

__device__ __forceinline__ v8f wmma_bf16(v16bf a, v16bf b, v8f c) {
  return __builtin_amdgcn_wmma_f32_16x16x32_bf16(false, a, false, b, (short)0, c, false, false);
}

__device__ __forceinline__ v8f vzero8() {
  v8f z = {0.f, 0.f, 0.f, 0.f, 0.f, 0.f, 0.f, 0.f};
  return z;
}

// ---------------------------------------------------------------------------
// Fused coupling layer:  x = m*x + (1-m)*(x*exp(s)+t),
// s = tanh(mlp_s(x*m)), t = mlp_t(x*m); mlp = lrelu(lrelu(xW1+b1)W2+b2)W3+b3
// Block: 256 threads (8 waves), 32 batch rows. Template H in {256, 512}.
// ---------------------------------------------------------------------------
template<int H>
__global__ __launch_bounds__(256)
void k_coupling(const float* __restrict__ xin, float* __restrict__ xout,
                int D, unsigned mask, NetPtrs SN, NetPtrs TN)
{
  constexpr int LDH = H + 8;       // LDS row stride (elems), keeps 16B align
  constexpr int CT  = H / 64;      // col-tiles per wave (4 col-groups of waves)

  __shared__ __align__(16) unsigned short xm_bf[32][32];
  __shared__ __align__(16) float          x_f[32][16];
  __shared__ __align__(16) unsigned short h1s[32][LDH];
  __shared__ __align__(16) unsigned short h2s[32][LDH];
  __shared__ __align__(16) float          o_part[4][32 * 16];
  __shared__ __align__(16) float          o_s[32 * 16];
  __shared__ __align__(16) float          o_t[32 * 16];

  const int tid  = threadIdx.x;
  const int lane = tid & 31, wave = tid >> 5;
  const int half = lane >> 4, l16 = lane & 15;
  const int row0 = blockIdx.x * 32;

  // Load x block; build masked bf16 input padded to K=32
  for (int i = tid; i < 32 * 16; i += 256) {
    int r = i >> 4, c = i & 15;
    float v = (c < D) ? xin[(size_t)(row0 + r) * D + c] : 0.f;
    x_f[r][c] = v;
    float xm = ((mask >> c) & 1u) ? v : 0.f;
    xm_bf[r][c]      = f2bf(xm);
    xm_bf[r][c + 16] = 0;
  }
  __syncthreads();

  for (int net = 0; net < 2; ++net) {
    NetPtrs P = net ? TN : SN;
    const int rt = wave >> 2;      // row-tile 0/1
    const int cw = wave & 3;       // col-group

    // ---- GEMM1: h1 = lrelu(xm @ W1 + b1)   (K = 32, single WMMA step)
    {
      v16bf a = loadA16(&xm_bf[rt * 16 + l16][0], 0, half);
      for (int t = 0; t < CT; ++t) {
        int nt = cw * CT + t;
        v16bf b = loadB16(P.W1T + (size_t)(nt * 16 + l16) * 32, 0, half);
        v8f acc = vzero8();
        acc = wmma_bf16(a, b, acc);
        int n = nt * 16 + l16;
        float bias = P.b1[n];
        #pragma unroll
        for (int v = 0; v < 8; ++v) {
          float x = acc[v] + bias;
          x = (x > 0.f) ? x : 0.01f * x;
          h1s[rt * 16 + 8 * half + v][n] = f2bf(x);
        }
      }
    }
    __syncthreads();

    // ---- GEMM2: h2 = lrelu(h1 @ W2 + b2)   (K = H; the hot loop)
    {
      v8f acc[CT];
      #pragma unroll
      for (int t = 0; t < CT; ++t) acc[t] = vzero8();
      for (int k = 0; k < H; k += 32) {
        v16bf a = loadA16(&h1s[rt * 16 + l16][0], k, half);
        #pragma unroll
        for (int t = 0; t < CT; ++t) {
          int nt = cw * CT + t;
          v16bf b = loadB16(P.W2T + (size_t)(nt * 16 + l16) * H, k, half);
          acc[t] = wmma_bf16(a, b, acc[t]);
        }
      }
      #pragma unroll
      for (int t = 0; t < CT; ++t) {
        int n = (cw * CT + t) * 16 + l16;
        float bias = P.b2[n];
        #pragma unroll
        for (int v = 0; v < 8; ++v) {
          float x = acc[t][v] + bias;
          x = (x > 0.f) ? x : 0.01f * x;
          h2s[rt * 16 + 8 * half + v][n] = f2bf(x);
        }
      }
    }
    __syncthreads();

    // ---- GEMM3: o = h2 @ W3  (N=16 padded; K split 4-ways across waves)
    {
      const int ks = wave & 3;
      constexpr int KS = H / 4;
      v8f acc = vzero8();
      for (int k = ks * KS; k < ks * KS + KS; k += 32) {
        v16bf a = loadA16(&h2s[rt * 16 + l16][0], k, half);
        v16bf b = loadB16(P.W3T + (size_t)l16 * H, k, half);
        acc = wmma_bf16(a, b, acc);
      }
      #pragma unroll
      for (int v = 0; v < 8; ++v)
        o_part[ks][(rt * 16 + 8 * half + v) * 16 + l16] = acc[v];
    }
    __syncthreads();

    // reduce partials + b3 (+tanh for s-net)
    float* o = net ? o_t : o_s;
    for (int i = tid; i < 32 * 16; i += 256) {
      float v = o_part[0][i] + o_part[1][i] + o_part[2][i] + o_part[3][i] + P.b3[i & 15];
      o[i] = net ? v : tanhf(v);
    }
    __syncthreads();
  }

  // coupling update
  for (int i = tid; i < 32 * 16; i += 256) {
    int r = i >> 4, c = i & 15;
    if (c < D) {
      float xv = x_f[r][c];
      bool  mk = ((mask >> c) & 1u) != 0;
      float s = o_s[i], t = o_t[i];
      float y = mk ? xv : (xv * __expf(s) + t);
      xout[(size_t)(row0 + r) * D + c] = y;
    }
  }
}

// ---------------------------------------------------------------------------
// Weight conversion: fp32 -> bf16, transpose to [n][k], pad K(W1)->32, N(W3)->16
// ---------------------------------------------------------------------------
__global__ void k_convert(const float* __restrict__ W1, const float* __restrict__ W2,
                          const float* __restrict__ W3, const float* __restrict__ b1,
                          const float* __restrict__ b2, const float* __restrict__ b3,
                          int D, int H, NetPtrs P)
{
  int total = H * 32 + H * H + 16 * H + 2 * H + 16;
  for (int i = blockIdx.x * blockDim.x + threadIdx.x; i < total; i += gridDim.x * blockDim.x) {
    int j = i;
    if (j < H * 32) { int n = j >> 5, k = j & 31;
      P.W1T[j] = (k < D) ? f2bf(W1[k * H + n]) : (unsigned short)0; continue; }
    j -= H * 32;
    if (j < H * H) { int n = j / H, k = j - n * H;
      P.W2T[j] = f2bf(W2[k * H + n]); continue; }
    j -= H * H;
    if (j < 16 * H) { int n = j / H, k = j - n * H;
      P.W3T[j] = (n < D) ? f2bf(W3[k * D + n]) : (unsigned short)0; continue; }
    j -= 16 * H;
    if (j < H) { P.b1[j] = b1[j]; continue; }
    j -= H;
    if (j < H) { P.b2[j] = b2[j]; continue; }
    j -= H;
    P.b3[j] = (j < D) ? b3[j] : 0.f;
  }
}

// _own(alpha1): closed-form 2x2 SPD inverse square root, a1 = S^{-1/2} @ Wc
__global__ void k_own(const float* __restrict__ alpha1, float* __restrict__ a1o)
{
  if (threadIdx.x != 0 || blockIdx.x != 0) return;
  float w0[10], w1[10], m0 = 0.f, m1 = 0.f;
  for (int j = 0; j < 10; ++j) { w0[j] = alpha1[j]; w1[j] = alpha1[10 + j]; m0 += w0[j]; m1 += w1[j]; }
  m0 *= 0.1f; m1 *= 0.1f;
  float s00 = 1e-5f, s01 = 0.f, s11 = 1e-5f;
  for (int j = 0; j < 10; ++j) {
    w0[j] -= m0; w1[j] -= m1;
    s00 += w0[j] * w0[j]; s01 += w0[j] * w1[j]; s11 += w1[j] * w1[j];
  }
  float tr = s00 + s11, det = s00 * s11 - s01 * s01;
  float sd = sqrtf(det), denom = sqrtf(tr + 2.f * sd);
  float q00 = (s00 + sd) / denom, q01 = s01 / denom, q11 = (s11 + sd) / denom; // sqrtm(S)
  float dq = q00 * q11 - q01 * q01;
  float i00 = q11 / dq, i01 = -q01 / dq, i11 = q00 / dq;                       // S^{-1/2}
  for (int j = 0; j < 10; ++j) {
    a1o[j]      = i00 * w0[j] + i01 * w1[j];
    a1o[10 + j] = i01 * w0[j] + i11 * w1[j];
  }
}

// bias = Z @ a1_own + alpha1_bias  (K=2, trivial)
__global__ void k_bias(const float* __restrict__ Z, const float* __restrict__ a1,
                       const float* __restrict__ ab, float* __restrict__ xo, int B)
{
  int r = blockIdx.x * blockDim.x + threadIdx.x;
  if (r >= B) return;
  float z0 = Z[2 * r], z1 = Z[2 * r + 1];
  #pragma unroll
  for (int j = 0; j < 10; ++j)
    xo[(size_t)r * 10 + j] = z0 * a1[j] + z1 * a1[10 + j] + ab[j];
}

__global__ void k_addslice(const float* __restrict__ A, int ldA, int offA,
                           const float* __restrict__ Bp, int ldB, int offB,
                           float* __restrict__ O, int ldO, int offO, int ncol, int nrow)
{
  int i = blockIdx.x * blockDim.x + threadIdx.x;
  if (i >= nrow * ncol) return;
  int r = i / ncol, c = i - r * ncol;
  O[(size_t)r * ldO + offO + c] = A[(size_t)r * ldA + offA + c] + Bp[(size_t)r * ldB + offB + c];
}

__global__ void k_copyslice(const float* __restrict__ A, int ldA, int offA,
                            float* __restrict__ O, int ldO, int offO, int ncol, int nrow)
{
  int i = blockIdx.x * blockDim.x + threadIdx.x;
  if (i >= nrow * ncol) return;
  int r = i / ncol, c = i - r * ncol;
  O[(size_t)r * ldO + offO + c] = A[(size_t)r * ldA + offA + c];
}

// ---------------------------------------------------------------------------
// Host orchestration.  d_in pytree flatten order (dicts alphabetical, lists
// in order): X, Z, alpha1, alpha1_bias, then per flow: s-nets then t-nets,
// each net as W1,W2,W3,b1,b2,b3.
// ---------------------------------------------------------------------------
extern "C" void kernel_launch(void* const* d_in, const int* in_sizes, int n_in,
                              void* d_out, int out_size, void* d_ws, size_t ws_size,
                              hipStream_t stream)
{
  (void)in_sizes; (void)n_in; (void)out_size; (void)ws_size;
  const int B = 32768;
  const float* X   = (const float*)d_in[0];
  const float* Z   = (const float*)d_in[1];
  const float* A1  = (const float*)d_in[2];
  const float* A1b = (const float*)d_in[3];
  float* out = (float*)d_out;

  size_t off = 0;
  char*  base = (char*)d_ws;
  auto alloc = [&](size_t bytes) -> void* {
    void* p = base + off;
    off = (off + bytes + 255) & ~(size_t)255;
    return p;
  };

  float* xA  = (float*)alloc((size_t)B * 10 * 4);
  float* va  = (float*)alloc((size_t)B * 2 * 4);
  float* vb  = (float*)alloc((size_t)B * 2 * 4);
  float* vc  = (float*)alloc((size_t)B * 2 * 4);
  float* vd  = (float*)alloc((size_t)B * 2 * 4);
  float* ve  = (float*)alloc((size_t)B * 2 * 4);
  float* vf  = (float*)alloc((size_t)B * 2 * 4);
  float* vg  = (float*)alloc((size_t)B * 2 * 4);
  float* vh  = (float*)alloc((size_t)B * 2 * 4);
  float* t2  = (float*)alloc((size_t)B * 2 * 4);
  float* ef4 = (float*)alloc((size_t)B * 4 * 4);
  float* t4  = (float*)alloc((size_t)B * 4 * 4);
  float* a1o = (float*)alloc(32 * 4);

  auto conv = [&](int ii, int D, int H) -> NetPtrs {
    NetPtrs p;
    p.W1T = (unsigned short*)alloc((size_t)H * 32 * 2);
    p.W2T = (unsigned short*)alloc((size_t)H * H * 2);
    p.W3T = (unsigned short*)alloc((size_t)16 * H * 2);
    p.b1  = (float*)alloc((size_t)H * 4);
    p.b2  = (float*)alloc((size_t)H * 4);
    p.b3  = (float*)alloc(16 * 4);
    int total = H * 32 + H * H + 16 * H + 2 * H + 16;
    int blocks = (total + 255) / 256;
    k_convert<<<blocks, 256, 0, stream>>>(
        (const float*)d_in[ii + 0], (const float*)d_in[ii + 1], (const float*)d_in[ii + 2],
        (const float*)d_in[ii + 3], (const float*)d_in[ii + 4], (const float*)d_in[ii + 5],
        D, H, p);
    return p;
  };

  // alpha flow (H=512, D=10, L=8): inputs 4..99 (48 s arrays then 48 t arrays)
  NetPtrs aS[8], aT[8];
  for (int l = 0; l < 8; ++l) {
    aS[l] = conv(4 + 6 * l, 10, 512);
    aT[l] = conv(4 + 48 + 6 * l, 10, 512);
  }
  // 2D/4D flows (H=256, L=2): ac,be,cf,dh,hx,gx,ef at 24-array strides
  const int fbase[7] = {100, 124, 148, 172, 196, 220, 244};
  const int fD[7]    = {2, 2, 2, 2, 2, 2, 4};
  NetPtrs fS[7][2], fT[7][2];
  for (int f = 0; f < 7; ++f)
    for (int l = 0; l < 2; ++l) {
      fS[f][l] = conv(fbase[f] + 6 * l,      fD[f], 256);
      fT[f][l] = conv(fbase[f] + 12 + 6 * l, fD[f], 256);
    }

  k_own<<<1, 1, 0, stream>>>(A1, a1o);
  k_bias<<<(B + 255) / 256, 256, 0, stream>>>(Z, a1o, A1b, xA, B);

  // alpha flow, in place on xA (masks: [1x5,0x5] / complement)
  for (int l = 0; l < 8; ++l) {
    unsigned m = (l & 1) ? 0x3E0u : 0x01Fu;
    k_coupling<512><<<B / 32, 256, 0, stream>>>(xA, xA, 10, m, aS[l], aT[l]);
  }

  auto flow2 = [&](int f, const float* src, float* dst, int D, unsigned m0, unsigned m1) {
    k_coupling<256><<<B / 32, 256, 0, stream>>>(src, dst, D, m0, fS[f][0], fT[f][0]);
    k_coupling<256><<<B / 32, 256, 0, stream>>>(dst, dst, D, m1, fS[f][1], fT[f][1]);
  };
  auto adds = [&](const float* A, int ldA, int offA, const float* Bp, int ldB, int offB,
                  float* O, int ldO, int offO, int ncol) {
    k_addslice<<<(unsigned)((B * ncol + 255) / 256), 256, 0, stream>>>(
        A, ldA, offA, Bp, ldB, offB, O, ldO, offO, ncol, B);
  };
  auto cps = [&](const float* A, int ldA, int offA, float* O, int ldO, int offO, int ncol) {
    k_copyslice<<<(unsigned)((B * ncol + 255) / 256), 256, 0, stream>>>(
        A, ldA, offA, O, ldO, offO, ncol, B);
  };

  enum { AC = 0, BE = 1, CF = 2, DH = 3, HX = 4, GX = 5, EF = 6 };

  adds(X, 10, 0, xA, 10, 0, va, 2, 0, 2);        // a  = X12 + b1
  adds(va, 2, 0, xA, 10, 2, vb, 2, 0, 2);        // b  = a + b2
  flow2(AC, va, t2, 2, 1u, 2u);
  adds(X, 10, 2, t2, 2, 0, vc, 2, 0, 2);         // c  = X34 + flow_ac(a)
  adds(vb, 2, 0, xA, 10, 4, vd, 2, 0, 2);        // d  = b + b3
  flow2(BE, vb, t2, 2, 1u, 2u);
  adds(vc, 2, 0, t2, 2, 0, ve, 2, 0, 2);         // e  = c + flow_be(b)
  flow2(CF, vc, t2, 2, 1u, 2u);
  adds(X, 10, 4, t2, 2, 0, vf, 2, 0, 2);         // f  = X56 + flow_cf(c)
  cps(ve, 2, 0, ef4, 4, 0, 2);                   // ef = concat(e, f)
  cps(vf, 2, 0, ef4, 4, 2, 2);
  adds(vd, 2, 0, xA, 10, 6, vg, 2, 0, 2);        // g  = d + b4
  flow2(DH, vd, t2, 2, 1u, 2u);
  adds(ve, 2, 0, t2, 2, 0, vh, 2, 0, 2);         // h  = e + flow_dh(d)
  adds(vg, 2, 0, xA, 10, 8, out, 10, 0, 2);      // Y12 = g + b5
  flow2(GX, vg, t2, 2, 1u, 2u);
  adds(vh, 2, 0, t2, 2, 0, out, 10, 2, 2);       // Y34 = h + flow_gx(g)
  flow2(HX, vh, t2, 2, 1u, 2u);
  adds(vf, 2, 0, t2, 2, 0, out, 10, 4, 2);       // Y56 = f + flow_hx(h)
  flow2(EF, ef4, t4, 4, 3u, 0xCu);
  adds(X, 10, 6, t4, 4, 0, out, 10, 6, 4);       // Y710 = X710 + flow_ef(ef)
}
